// Pointnet2MSGSEM_86792699117873
// MI455X (gfx1250) — matrix operand order repaired and checked by reading, not compiled, not run on hardware
//
#include <hip/hip_runtime.h>
#include <cstddef>
#include <cstdint>

typedef __attribute__((ext_vector_type(16))) _Float16 v16h;
typedef __attribute__((ext_vector_type(8)))  _Float16 v8h;
typedef __attribute__((ext_vector_type(8)))  float    v8f;

#define TPB 256

// ------------------------------------------------------------------
// Problem constants (match reference)
// ------------------------------------------------------------------
static constexpr int BB  = 2;
static constexpr int NN0 = 16384;
static constexpr int NP0 = 4096, NP1 = 1024, NP2 = 256;

static constexpr int rnd16c(int x) { return (x + 15) & ~15; }
static constexpr int rnd32c(int x) { return (x + 31) & ~31; }

// ------------------------------------------------------------------
// Workspace plan.
// f32 region: coordinates, per-level feature maps, FP outputs, 3NN weights.
// f16 region: GEMM activations (padded ld), ping-pong H arenas, padded weights.
// int region: fps / ball-query / knn indices.
// ------------------------------------------------------------------
static constexpr size_t F_XYZ1 = 0;
static constexpr size_t F_XYZ2 = F_XYZ1 + (size_t)BB*NP0*3;
static constexpr size_t F_XYZ3 = F_XYZ2 + (size_t)BB*NP1*3;
static constexpr size_t F_F1   = F_XYZ3 + (size_t)BB*NP2*3;          // (B,4096,128)
static constexpr size_t F_F2   = F_F1   + (size_t)BB*NP0*128;        // (B,1024,512)
static constexpr size_t F_F3   = F_F2   + (size_t)BB*NP1*512;        // (B,256,1024)
static constexpr size_t F_FP2O = F_F3   + (size_t)BB*NP2*1024;       // (B,1024,512)
static constexpr size_t F_FP1O = F_FP2O + (size_t)BB*NP1*512;        // (B,4096,256)
static constexpr size_t F_NNW  = F_FP1O + (size_t)BB*NP0*256;        // (B,16384,3)
static constexpr size_t F32_END = ((F_NNW + (size_t)BB*NN0*3) + 3) & ~(size_t)3;

// halves, offsets from the f16 base
static constexpr size_t H_X   = 0;
static constexpr size_t XH_SZ = 10485760;   // >= 65536*160 (SA1 s1), 16384*544 (SA2 s1)
static constexpr size_t H_A   = H_X + XH_SZ;
static constexpr size_t HA_SZ = 16777216;   // == 262144*64
static constexpr size_t H_B   = H_A + HA_SZ;
static constexpr size_t H_W   = H_B + HA_SZ;
static constexpr size_t W_SZ  = 2621440;    // >= 2,482,176 padded weight halves
static constexpr size_t H_END = H_W + W_SZ;

struct GLayer { const _Float16* wt; int kpad; int n; const float* g; const float* b; };

static inline int cdiv(long long a, int b) { return (int)((a + b - 1) / b); }

// ------------------------------------------------------------------
// GEMM: Y = epilogue(X(M x K) * W(K x N)) via V_WMMA_F32_16X16X32_F16.
//  X : f16, row stride lda (= Kpad32), rows fully valid (M % 16 == 0)
//  Wt: f16, transposed + padded, Wt[n*ldb + k], ldb = Kpad32, Npad rows
//  out: f16 (padded ld, zero-filled cols) or f32 (masked store, stride out_ld)
// All hot-loop loads are unconditional, 16B-aligned b128 loads.
// ------------------------------------------------------------------
__global__ __launch_bounds__(TPB) void wmma_gemm_kernel(
    const _Float16* __restrict__ X, int lda,
    const _Float16* __restrict__ Wt, int ldb,
    const float* __restrict__ G, const float* __restrict__ Bv,
    _Float16* __restrict__ Yh, float* __restrict__ Yf,
    int M, int N, int out_ld, int relu)
{
  const int wave = threadIdx.x >> 5;
  const int lane = threadIdx.x & 31;
  const int m0 = (blockIdx.y * 8 + wave) * 16;
  const int n0 = blockIdx.x * 16;
  if (m0 >= M) return;                            // uniform per wave

  const int akoff = (lane >> 4) * 8;              // A: K sub-group
  const int bkoff = (lane >> 4) * 8;              // B: K sub-group
  const _Float16* xrow = X  + (size_t)(m0 + (lane & 15)) * lda + akoff;
  const _Float16* wrow = Wt + (size_t)(n0 + (lane & 15)) * ldb + bkoff;

  v8f acc = {0.f,0.f,0.f,0.f,0.f,0.f,0.f,0.f};
  for (int k0 = 0; k0 < lda; k0 += 32) {
    const v8h a0 = *(const v8h*)(xrow + k0);
    const v8h a1 = *(const v8h*)(xrow + k0 + 16);
    const v8h b0 = *(const v8h*)(wrow + k0);
    const v8h b1 = *(const v8h*)(wrow + k0 + 16);
    const v16h a = __builtin_shufflevector(a0, a1, 0,1,2,3,4,5,6,7,8,9,10,11,12,13,14,15);
    const v16h b = __builtin_shufflevector(b0, b1, 0,1,2,3,4,5,6,7,8,9,10,11,12,13,14,15);
    acc = __builtin_amdgcn_wmma_f32_16x16x32_f16(false, a, false, b,
                                                 (short)0, acc, false, false);
  }

  const int bn = n0 + (lane & 15);
  const bool nok = bn < N;
  const float gv = (G  && nok) ? G[bn]  : 1.f;
  const float bv = (Bv && nok) ? Bv[bn] : 0.f;
#pragma unroll
  for (int r = 0; r < 8; ++r) {
    const int m = m0 + r + 8 * (lane >> 4);
    float v = acc[r] * gv + bv;
    if (relu) v = fmaxf(v, 0.f);
    if (Yh) {
      Yh[(size_t)m * out_ld + bn] = (_Float16)(nok ? v : 0.f);
    } else if (nok) {
      Yf[(size_t)m * out_ld + bn] = v;
    }
  }
}

// Pad + transpose + f16-convert one weight matrix: Wt[n*Kpad+k] = W[k*N+n]
__global__ void pad_weight_kernel(const float* __restrict__ W, _Float16* __restrict__ Wt,
                                  int K, int N, int Kpad, int Npad)
{
  const int t = blockIdx.x * blockDim.x + threadIdx.x;
  const long long total = (long long)Npad * Kpad;
  if (t >= total) return;
  const int k = t % Kpad;
  const int n = t / Kpad;
  const float v = (n < N && k < K) ? W[(size_t)k * N + n] : 0.f;
  Wt[t] = (_Float16)v;
}

// ------------------------------------------------------------------
// FPS: one block per batch, dmin in registers, LDS argmax tree
// with lowest-index tie-break (matches jnp.argmax semantics).
// ------------------------------------------------------------------
template <int PPT>
__global__ __launch_bounds__(TPB) void fps_kernel(const float* __restrict__ xyz,
                                                  int* __restrict__ fidx,
                                                  int N, int npoint)
{
  __shared__ float sval[TPB];
  __shared__ int   sidx[TPB];
  const int b = blockIdx.x;
  const float* P = xyz + (size_t)b * N * 3;
  float dmin[PPT];
#pragma unroll
  for (int i = 0; i < PPT; ++i) dmin[i] = 1e10f;
  int far = 0;
  for (int it = 0; it < npoint; ++it) {
    if (threadIdx.x == 0) fidx[(size_t)b * npoint + it] = far;
    const float cx = P[far * 3 + 0];
    const float cy = P[far * 3 + 1];
    const float cz = P[far * 3 + 2];
    float best = -1.f; int bi = 0;
#pragma unroll
    for (int i = 0; i < PPT; ++i) {
      const int j = threadIdx.x + i * TPB;
      const float dx = P[j * 3 + 0] - cx;
      const float dy = P[j * 3 + 1] - cy;
      const float dz = P[j * 3 + 2] - cz;
      const float d  = dx * dx + dy * dy + dz * dz;
      dmin[i] = fminf(dmin[i], d);
      if (dmin[i] > best) { best = dmin[i]; bi = j; }
    }
    sval[threadIdx.x] = best; sidx[threadIdx.x] = bi;
    __syncthreads();
    for (int off = TPB >> 1; off > 0; off >>= 1) {
      if (threadIdx.x < off) {
        const float v2 = sval[threadIdx.x + off];
        const int   i2 = sidx[threadIdx.x + off];
        if (v2 > sval[threadIdx.x] ||
            (v2 == sval[threadIdx.x] && i2 < sidx[threadIdx.x])) {
          sval[threadIdx.x] = v2; sidx[threadIdx.x] = i2;
        }
      }
      __syncthreads();
    }
    far = sidx[0];
    __syncthreads();
  }
}

__global__ void gather_xyz_kernel(const float* __restrict__ xyz, const int* __restrict__ fidx,
                                  float* __restrict__ nxyz, int B, int N, int np)
{
  const int t = blockIdx.x * blockDim.x + threadIdx.x;
  if (t >= B * np * 3) return;
  const int c = t % 3;
  const int i = t / 3;
  const int b = i / np;
  nxyz[t] = xyz[((size_t)b * N + fidx[i]) * 3 + c];
}

// Ball query: first `ns` indices (ascending) within radius; pad with first (or 0).
__global__ void ball_query_kernel(const float* __restrict__ xyz, const float* __restrict__ nxyz,
                                  int* __restrict__ gidx, int B, int N, int np, int ns, float r2)
{
  const int t = blockIdx.x * blockDim.x + threadIdx.x;
  if (t >= B * np) return;
  const int b = t / np;
  const float qx = nxyz[(size_t)t * 3 + 0];
  const float qy = nxyz[(size_t)t * 3 + 1];
  const float qz = nxyz[(size_t)t * 3 + 2];
  int* out = gidx + (size_t)t * ns;
  const float* P = xyz + (size_t)b * N * 3;
  int cnt = 0;
  for (int j = 0; j < N && cnt < ns; ++j) {
    const float dx = P[j * 3 + 0] - qx;
    const float dy = P[j * 3 + 1] - qy;
    const float dz = P[j * 3 + 2] - qz;
    if (dx * dx + dy * dy + dz * dz < r2) out[cnt++] = j;
  }
  const int first = (cnt > 0) ? out[0] : 0;
  for (int s = cnt; s < ns; ++s) out[s] = first;
}

// Build grouped GEMM input (f16, padded lda): [0:3)=xyz-new_xyz, [3:3+Cf)=feats, rest 0
__global__ void group_kernel(const float* __restrict__ xyz, const float* __restrict__ nxyz,
                             const float* __restrict__ feats, const int* __restrict__ gidx,
                             _Float16* __restrict__ X, int B, int N, int np, int ns,
                             int Cf, int lda)
{
  const long long total = (long long)B * np * ns * lda;
  const long long t = (long long)blockIdx.x * blockDim.x + threadIdx.x;
  if (t >= total) return;
  const int c = (int)(t % lda);
  const long long row = t / lda;
  const long long bi = row / ns;
  const int i = (int)(bi % np);
  const int b = (int)(bi / np);
  const int j = gidx[row];
  float v = 0.f;
  if (c < 3)
    v = xyz[((size_t)b * N + j) * 3 + c] - nxyz[((size_t)b * np + i) * 3 + c];
  else if (c < 3 + Cf)
    v = feats[((size_t)b * N + j) * Cf + (c - 3)];
  X[t] = (_Float16)v;
}

// Max over nsample (f16 H, padded ldH) -> f32 concat slot of level feature map.
__global__ void maxpool_kernel(const _Float16* __restrict__ H, float* __restrict__ F,
                               int npts, int ns, int C, int ldH, int Ctot, int coff)
{
  const int t = blockIdx.x * blockDim.x + threadIdx.x;
  if (t >= npts * C) return;
  const int c = t % C;
  const int p = t / C;
  float m = -3.4e38f;
  for (int s = 0; s < ns; ++s)
    m = fmaxf(m, (float)H[((size_t)p * ns + s) * ldH + c]);
  F[(size_t)p * Ctot + coff + c] = m;
}

// 3-NN (smallest d2, lowest-index ties) + normalized inverse-distance weights.
__global__ void knn3_kernel(const float* __restrict__ unk, const float* __restrict__ kn,
                            int* __restrict__ nni, float* __restrict__ nnw,
                            int B, int Nu, int Nk)
{
  const int t = blockIdx.x * blockDim.x + threadIdx.x;
  if (t >= B * Nu) return;
  const int b = t / Nu;
  const float qx = unk[(size_t)t * 3 + 0];
  const float qy = unk[(size_t)t * 3 + 1];
  const float qz = unk[(size_t)t * 3 + 2];
  const float* P = kn + (size_t)b * Nk * 3;
  float d0 = 3.4e38f, d1 = 3.4e38f, d2 = 3.4e38f;
  int   i0 = 0, i1 = 0, i2 = 0;
  for (int j = 0; j < Nk; ++j) {
    const float dx = P[j * 3 + 0] - qx;
    const float dy = P[j * 3 + 1] - qy;
    const float dz = P[j * 3 + 2] - qz;
    const float d  = dx * dx + dy * dy + dz * dz;
    if (d < d0)      { d2 = d1; i2 = i1; d1 = d0; i1 = i0; d0 = d; i0 = j; }
    else if (d < d1) { d2 = d1; i2 = i1; d1 = d;  i1 = j; }
    else if (d < d2) { d2 = d;  i2 = j; }
  }
  float w0 = 1.f / (d0 + 1e-8f);
  float w1 = 1.f / (d1 + 1e-8f);
  float w2 = 1.f / (d2 + 1e-8f);
  const float inv = 1.f / (w0 + w1 + w2);
  nni[(size_t)t * 3 + 0] = i0; nni[(size_t)t * 3 + 1] = i1; nni[(size_t)t * 3 + 2] = i2;
  nnw[(size_t)t * 3 + 0] = w0 * inv; nnw[(size_t)t * 3 + 1] = w1 * inv; nnw[(size_t)t * 3 + 2] = w2 * inv;
}

// FP GEMM input (f16, padded lda): [0:Ck)=interp(known feats), [Ck:Ck+Cs)=skip, rest 0
__global__ void fp_build_kernel(const int* __restrict__ nni, const float* __restrict__ nnw,
                                const float* __restrict__ kf, const float* __restrict__ skip,
                                _Float16* __restrict__ X, int B, int Nu, int Nk,
                                int Ck, int Cs, int lda)
{
  const long long total = (long long)B * Nu * lda;
  const long long t = (long long)blockIdx.x * blockDim.x + threadIdx.x;
  if (t >= total) return;
  const int c = (int)(t % lda);
  const long long row = t / lda;
  const int b = (int)(row / Nu);
  float v = 0.f;
  if (c < Ck) {
#pragma unroll
    for (int j = 0; j < 3; ++j) {
      const int   idx = nni[row * 3 + j];
      const float w   = nnw[row * 3 + j];
      v += w * kf[((size_t)b * Nk + idx) * Ck + c];
    }
  } else if (c < Ck + Cs) {
    v = skip[(size_t)row * Cs + (c - Ck)];
  }
  X[t] = (_Float16)v;
}

// ------------------------------------------------------------------
// Host-side helpers
// ------------------------------------------------------------------
static inline void gemm_h16(hipStream_t st, const _Float16* X, const GLayer& L,
                            _Float16* Yh, int M, bool relu)
{
  const int out_ld = rnd32c(L.n);
  dim3 grid((unsigned)(out_ld / 16), (unsigned)cdiv(M, 128));
  wmma_gemm_kernel<<<grid, TPB, 0, st>>>(X, L.kpad, L.wt, L.kpad, L.g, L.b,
                                         Yh, nullptr, M, L.n, out_ld, relu ? 1 : 0);
}
static inline void gemm_f32(hipStream_t st, const _Float16* X, const GLayer& L,
                            float* Yf, int M, bool relu)
{
  dim3 grid((unsigned)cdiv(L.n, 16), (unsigned)cdiv(M, 128));
  wmma_gemm_kernel<<<grid, TPB, 0, st>>>(X, L.kpad, L.wt, L.kpad, L.g, L.b,
                                         nullptr, Yf, M, L.n, L.n, relu ? 1 : 0);
}

static void run_sa(hipStream_t st, const float* xyz, int N, const float* feats, int Cf,
                   int np, const float* radii, const int* nsamp, const GLayer L[2][3],
                   float* nxyz, float* fout, int Ctot,
                   int* fidx, int* gidx, _Float16* Xh, _Float16* Ha, _Float16* Hb)
{
  if (N == 16384)      fps_kernel<64><<<BB, TPB, 0, st>>>(xyz, fidx, N, np);
  else if (N == 4096)  fps_kernel<16><<<BB, TPB, 0, st>>>(xyz, fidx, N, np);
  else                 fps_kernel< 4><<<BB, TPB, 0, st>>>(xyz, fidx, N, np);
  gather_xyz_kernel<<<cdiv((long long)BB * np * 3, TPB), TPB, 0, st>>>(xyz, fidx, nxyz, BB, N, np);

  const int lda = rnd32c(3 + Cf);
  for (int s = 0; s < 2; ++s) {
    const int   ns = nsamp[s];
    const float r2 = radii[s] * radii[s];
    const int   M  = BB * np * ns;
    ball_query_kernel<<<cdiv((long long)BB * np, TPB), TPB, 0, st>>>(xyz, nxyz, gidx, BB, N, np, ns, r2);
    group_kernel<<<cdiv((long long)M * lda, TPB), TPB, 0, st>>>(xyz, nxyz, feats, gidx, Xh, BB, N, np, ns, Cf, lda);
    gemm_h16(st, Xh, L[s][0], Ha, M, true);
    gemm_h16(st, Ha, L[s][1], Hb, M, true);
    gemm_h16(st, Hb, L[s][2], Ha, M, true);
    const int C = L[s][2].n;
    maxpool_kernel<<<cdiv((long long)BB * np * C, TPB), TPB, 0, st>>>(Ha, fout, BB * np, ns, C, rnd32c(C), Ctot, s * C);
  }
}

// ------------------------------------------------------------------
// Entry point
// ------------------------------------------------------------------
extern "C" void kernel_launch(void* const* d_in, const int* in_sizes, int n_in,
                              void* d_out, int out_size, void* d_ws, size_t ws_size,
                              hipStream_t stream)
{
  (void)in_sizes; (void)n_in; (void)out_size; (void)ws_size;

  const float* pc = (const float*)d_in[0];           // (B, 16384, 3)

  // ---- workspace carve ----
  float* ws   = (float*)d_ws;
  float* xyz1 = ws + F_XYZ1;
  float* xyz2 = ws + F_XYZ2;
  float* xyz3 = ws + F_XYZ3;
  float* f1   = ws + F_F1;
  float* f2   = ws + F_F2;
  float* f3   = ws + F_F3;
  float* fp2o = ws + F_FP2O;
  float* fp1o = ws + F_FP1O;
  float* nnw  = ws + F_NNW;

  _Float16* hb_base = (_Float16*)(ws + F32_END);
  _Float16* Xh = hb_base + H_X;
  _Float16* Ha = hb_base + H_A;
  _Float16* Hb = hb_base + H_B;
  _Float16* wp = hb_base + H_W;

  int* iws  = (int*)(hb_base + H_END);
  int* fidx = iws;                                   // B*4096
  int* gidx = fidx + (size_t)BB * NP0;               // B*4096*32
  int* nni  = gidx + (size_t)BB * NP0 * 32;          // B*16384*3

  // ---- parameter tensors in setup_inputs() insertion order ----
  // d_in[1..]: sa[3][2][3]{w,g,b}, fp[3][2]{w,g,b}, fc{w1,g1,b1,w2,b2}
  // Pre-pad/transpose/convert every weight into the f16 arena.
  _Float16* wcur = wp;
  auto prep = [&](int pi, int K, int N) -> GLayer {
    const float* w = (const float*)d_in[pi + 0];
    const float* g = (const float*)d_in[pi + 1];
    const float* b = (const float*)d_in[pi + 2];
    const int Kp = rnd32c(K), Np = rnd32c(N);
    pad_weight_kernel<<<cdiv((long long)Np * Kp, TPB), TPB, 0, stream>>>(w, wcur, K, N, Kp, Np);
    GLayer L{wcur, Kp, N, g, b};
    wcur += (size_t)Np * Kp;
    return L;
  };

  static const int SA_CH[3][4] = {{3,45,45,64},{131,192,192,256},{515,320,320,512}};
  static const int FP_CH[3][3] = {{256,128,128},{640,256,256},{1536,512,512}};

  GLayer sa[3][2][3];
  int p = 1;
  for (int k = 0; k < 3; ++k)
    for (int s = 0; s < 2; ++s)
      for (int l = 0; l < 3; ++l) { sa[k][s][l] = prep(p, SA_CH[k][l], SA_CH[k][l+1]); p += 3; }
  GLayer fpl[3][2];
  for (int k = 0; k < 3; ++k)
    for (int l = 0; l < 2; ++l) { fpl[k][l] = prep(p, FP_CH[k][l], FP_CH[k][l+1]); p += 3; }

  // fc: w1,g1,b1,w2,b2 — pad via the same path (fc2 has no g: pass nullptr)
  GLayer fc1 = prep(p, 128, 128);                     // w1,g1,b1
  const float* fc_w2 = (const float*)d_in[p + 3];
  const float* fc_b2 = (const float*)d_in[p + 4];
  const int K2p = rnd32c(128), N2p = rnd32c(13);
  pad_weight_kernel<<<cdiv((long long)N2p * K2p, TPB), TPB, 0, stream>>>(fc_w2, wcur, 128, 13, K2p, N2p);
  GLayer fc2{wcur, K2p, 13, nullptr, fc_b2};
  wcur += (size_t)N2p * K2p;

  // ---------------- Set abstraction ----------------
  {
    const float r[2] = {0.4f, 0.8f}; const int nss[2] = {16, 32};
    run_sa(stream, pc, NN0, nullptr, 0, NP0, r, nss, sa[0], xyz1, f1, 128, fidx, gidx, Xh, Ha, Hb);
  }
  {
    const float r[2] = {0.8f, 1.2f}; const int nss[2] = {16, 32};
    run_sa(stream, xyz1, NP0, f1, 128, NP1, r, nss, sa[1], xyz2, f2, 512, fidx, gidx, Xh, Ha, Hb);
  }
  {
    const float r[2] = {1.2f, 1.6f}; const int nss[2] = {16, 32};
    run_sa(stream, xyz2, NP1, f2, 512, NP2, r, nss, sa[2], xyz3, f3, 1024, fidx, gidx, Xh, Ha, Hb);
  }

  // ---------------- Feature propagation ----------------
  // FP2: unknown=xyz2(1024), known=xyz3(256), kf=f3(1024), skip=f2(512) -> fp2o f32
  {
    knn3_kernel<<<cdiv((long long)BB * NP1, TPB), TPB, 0, stream>>>(xyz2, xyz3, nni, nnw, BB, NP1, NP2);
    const int lda = rnd32c(1024 + 512);
    fp_build_kernel<<<cdiv((long long)BB * NP1 * lda, TPB), TPB, 0, stream>>>(nni, nnw, f3, f2, Xh, BB, NP1, NP2, 1024, 512, lda);
    gemm_h16(stream, Xh, fpl[2][0], Ha, BB * NP1, true);
    gemm_f32(stream, Ha, fpl[2][1], fp2o, BB * NP1, true);
  }
  // FP1: unknown=xyz1(4096), known=xyz2(1024), kf=fp2o(512), skip=f1(128) -> fp1o f32
  {
    knn3_kernel<<<cdiv((long long)BB * NP0, TPB), TPB, 0, stream>>>(xyz1, xyz2, nni, nnw, BB, NP0, NP1);
    const int lda = rnd32c(512 + 128);
    fp_build_kernel<<<cdiv((long long)BB * NP0 * lda, TPB), TPB, 0, stream>>>(nni, nnw, fp2o, f1, Xh, BB, NP0, NP1, 512, 128, lda);
    gemm_h16(stream, Xh, fpl[1][0], Ha, BB * NP0, true);
    gemm_f32(stream, Ha, fpl[1][1], fp1o, BB * NP0, true);
  }
  // FP0: unknown=pc(16384), known=xyz1(4096), kf=fp1o(256), no skip -> f16 chain into FC
  const int Mfc = BB * NN0;
  {
    knn3_kernel<<<cdiv((long long)BB * NN0, TPB), TPB, 0, stream>>>(pc, xyz1, nni, nnw, BB, NN0, NP0);
    const int lda = rnd32c(256);
    fp_build_kernel<<<cdiv((long long)BB * NN0 * lda, TPB), TPB, 0, stream>>>(nni, nnw, fp1o, nullptr, Xh, BB, NN0, NP0, 256, 0, lda);
    gemm_h16(stream, Xh, fpl[0][0], Ha, Mfc, true);   // 256 -> 128
    gemm_h16(stream, Ha, fpl[0][1], Hb, Mfc, true);   // 128 -> 128
  }

  // ---------------- FC head ----------------
  gemm_h16(stream, Hb, fc1, Ha, Mfc, true);           // 128 -> 128 (g1,b1,relu)
  gemm_f32(stream, Ha, fc2, (float*)d_out, Mfc, false); // 128 -> 13 (+b2)
}
